// learnableDAT_19335942766975
// MI455X (gfx1250) — matrix operand order repaired and checked by reading, not compiled, not run on hardware
//
#include <hip/hip_runtime.h>
#include <hip/hip_bf16.h>
#include <math.h>

#define B_    2
#define CIN   384
#define HIN   48
#define WIN   48
#define DIM_  768
#define HEADS_ 12
#define DH_   64
#define G_    12
#define HOUT  24
#define WOUT  24
#define NPOS  576          // 24*24
#define NQ_   (B_*NPOS)    // 1152
#define H2_   6
#define W2_   6
#define NJ_   36
#define CPBD  192
#define W1S   200          // padded LDS stride for W1^T (400B: 16 distinct banks)

typedef __attribute__((ext_vector_type(16))) _Float16 v16h;
typedef __attribute__((ext_vector_type(8)))  _Float16 v8h;
typedef __attribute__((ext_vector_type(8)))  float    v8f;
typedef __attribute__((ext_vector_type(4)))  unsigned int u32x4;
typedef __attribute__((ext_vector_type(8)))  int      i32x8;
typedef __attribute__((ext_vector_type(4)))  int      i32x4;

__device__ __forceinline__ v8f wmma_f16f32(v16h a, v16h b, v8f c) {
  return __builtin_amdgcn_wmma_f32_16x16x32_f16(false, a, false, b, (short)0, c,
                                                false, false);
}
// wave32 fragment index maps (ISA 7.12.2)
__device__ __forceinline__ int a_kidx(int e, int h) { return (e & 7) + ((e >> 3) << 4) + (h << 3); }
__device__ __forceinline__ int b_kidx(int e, int h) { return (e & 7) + ((e >> 3) << 4) + (h << 3); }

// ---------------------------------------------------------------------------
// 1) proj: 3x3 stride-2 conv as im2col GEMM with K permuted to (tap, ci):
//    per-tap scalar (ky,kx,iy,ix), contiguous ci => constant-offset loads.
// ---------------------------------------------------------------------------
__global__ __launch_bounds__(32) void proj_gemm(const float* __restrict__ x,
                                                const float* __restrict__ w,
                                                const float* __restrict__ bias,
                                                float* __restrict__ xp) {
  const int mo = blockIdx.x * 16;
  const int no = blockIdx.y * 16;
  const int lane = threadIdx.x & 31;
  const int h = lane >> 4, rr = lane & 15;
  const int n = no + rr;
  const int b = n / NPOS, p = n % NPOS;
  const int oy = p / WOUT, ox = p % WOUT;
  const float* wrow = w + (size_t)(mo + rr) * (CIN * 9);
  v8f acc = {};
  for (int tap = 0; tap < 9; ++tap) {
    const int ky = tap / 3, kx = tap % 3;
    const int iy = oy * 2 - 1 + ky, ix = ox * 2 - 1 + kx;
    const bool valid = (iy >= 0 && iy < HIN && ix >= 0 && ix < WIN);
    const float fm = valid ? 1.f : 0.f;
    // invalid lanes read a safe in-bounds address and are masked by fm
    const float* bbase = x + (size_t)b * CIN * HIN * WIN + (valid ? iy * WIN + ix : 0);
    for (int c0 = 0; c0 < CIN; c0 += 32) {
      v16h af, bf;
#pragma unroll
      for (int e = 0; e < 16; ++e) {
        af[e] = (_Float16)wrow[(size_t)(c0 + a_kidx(e, h)) * 9 + tap];
        bf[e] = (_Float16)(bbase[(size_t)(c0 + b_kidx(e, h)) * (HIN * WIN)] * fm);
      }
      acc = wmma_f16f32(af, bf, acc);
    }
  }
#pragma unroll
  for (int v = 0; v < 8; ++v) {
    int m = mo + v + h * 8;
    xp[((size_t)b * DIM_ + m) * NPOS + p] = acc[v] + bias[m];
  }
}

// ---------------------------------------------------------------------------
// 2) channel LayerNorm over NCHW (one wave per spatial position)
// ---------------------------------------------------------------------------
__global__ void ln_kernel(const float* __restrict__ in, float* __restrict__ out,
                          const float* __restrict__ g, const float* __restrict__ bt) {
  int wid = (blockIdx.x * blockDim.x + threadIdx.x) >> 5;
  int lane = threadIdx.x & 31;
  if (wid >= NQ_) return;
  int b = wid / NPOS, p = wid % NPOS;
  const float* base = in + (size_t)b * DIM_ * NPOS + p;
  float s = 0.f, s2 = 0.f;
  for (int c = lane; c < DIM_; c += 32) {
    float v = base[(size_t)c * NPOS];
    s += v; s2 += v * v;
  }
  for (int m = 16; m; m >>= 1) { s += __shfl_xor(s, m, 32); s2 += __shfl_xor(s2, m, 32); }
  float mu = s * (1.f / DIM_);
  float var = s2 * (1.f / DIM_) - mu * mu;
  float inv = rsqrtf(var + 1e-5f);
  float* ob = out + (size_t)b * DIM_ * NPOS + p;
  for (int c = lane; c < DIM_; c += 32) {
    float v = base[(size_t)c * NPOS];
    ob[(size_t)c * NPOS] = (v - mu) * inv * g[c] + bt[c];
  }
}

// ---------------------------------------------------------------------------
// 3) grouped 1x1 conv (q / k / v): per group 64x64xN GEMM (WMMA)
// ---------------------------------------------------------------------------
__global__ __launch_bounds__(32) void grouped_gemm(const float* __restrict__ src,
                                                   const float* __restrict__ w,
                                                   float* __restrict__ dst, int Np) {
  const int g = blockIdx.z;
  const int mo = blockIdx.x * 16, no = blockIdx.y * 16;
  const int lane = threadIdx.x & 31;
  const int h = lane >> 4, rr = lane & 15;
  const int N = B_ * Np;
  const int n = no + rr;
  const bool valid = (n < N);
  const int b = valid ? n / Np : 0;
  const int p = valid ? n % Np : 0;
  v8f acc = {};
  for (int kt = 0; kt < DH_; kt += 32) {
    v16h af, bf;
#pragma unroll
    for (int e = 0; e < 16; ++e) {
      af[e] = (_Float16)w[((size_t)g * DH_ + mo + rr) * DH_ + kt + a_kidx(e, h)];
      int kk = kt + b_kidx(e, h);
      float v = valid ? src[(((size_t)b * G_ + g) * DH_ + kk) * Np + p] : 0.f;
      bf[e] = (_Float16)v;
    }
    acc = wmma_f16f32(af, bf, acc);
  }
  if (valid) {
#pragma unroll
    for (int v = 0; v < 8; ++v) {
      int m = mo + v + h * 8;
      dst[(((size_t)b * G_ + g) * DH_ + m) * Np + p] = acc[v];
    }
  }
}

// ---------------------------------------------------------------------------
// 4) offset net: depthwise 6x6 stride-4 conv + exact GELU
// ---------------------------------------------------------------------------
__global__ void offconv_kernel(const float* __restrict__ q, const float* __restrict__ w,
                               const float* __restrict__ bias, float* __restrict__ h1) {
  int idx = blockIdx.x * blockDim.x + threadIdx.x;
  if (idx >= B_ * G_ * DH_ * NJ_) return;
  int p = idx % NJ_;
  int t = idx / NJ_;
  int c = t % DH_;
  int bg = t / DH_;
  int oy = p / W2_, ox = p % W2_;
  float acc = bias[c];
  const float* img = q + (size_t)(bg * DH_ + c) * NPOS;
#pragma unroll
  for (int ky = 0; ky < 6; ++ky) {
    int iy = oy * 4 - 1 + ky;
    if (iy < 0 || iy >= HOUT) continue;
#pragma unroll
    for (int kx = 0; kx < 6; ++kx) {
      int ix = ox * 4 - 1 + kx;
      if (ix < 0 || ix >= WOUT) continue;
      acc += img[iy * WOUT + ix] * w[c * 36 + ky * 6 + kx];
    }
  }
  h1[idx] = 0.5f * acc * (1.f + erff(acc * 0.70710678118654752f));   // exact GELU
}

// ---------------------------------------------------------------------------
// 5) offsets -> tanh*4 -> normalized sampling grid
// ---------------------------------------------------------------------------
__global__ void offsets_kernel(const float* __restrict__ h1, const float* __restrict__ w2,
                               float* __restrict__ vgrid) {
  int idx = blockIdx.x * blockDim.x + threadIdx.x;
  if (idx >= B_ * G_ * NJ_) return;
  int bg = idx / NJ_, p = idx % NJ_;
  int oy = p / W2_, ox = p % W2_;
  float a0 = 0.f, a1 = 0.f;
  for (int c = 0; c < DH_; ++c) {
    float hv = h1[(size_t)(bg * DH_ + c) * NJ_ + p];
    a0 += hv * w2[c];
    a1 += hv * w2[DH_ + c];
  }
  float vx = (float)ox + tanhf(a0) * 4.f;
  float vy = (float)oy + tanhf(a1) * 4.f;
  vgrid[idx * 2 + 0] = 2.f * vx / (float)(H2_ - 1) - 1.f;
  vgrid[idx * 2 + 1] = 2.f * vy / (float)(W2_ - 1) - 1.f;
}

// ---------------------------------------------------------------------------
// 6) bilinear grid sample (zeros padding, align_corners=False)
// ---------------------------------------------------------------------------
__global__ void gridsample_kernel(const float* __restrict__ xn,
                                  const float* __restrict__ vgrid,
                                  float* __restrict__ kv) {
  int idx = blockIdx.x * blockDim.x + threadIdx.x;
  if (idx >= B_ * DIM_ * NJ_) return;
  int p = idx % NJ_;
  int ch = idx / NJ_;        // = bg*64 + c
  int bg = ch >> 6;
  float gx = vgrid[(bg * NJ_ + p) * 2 + 0];
  float gy = vgrid[(bg * NJ_ + p) * 2 + 1];
  float xf = ((gx + 1.f) * (float)WOUT - 1.f) * 0.5f;
  float yf = ((gy + 1.f) * (float)HOUT - 1.f) * 0.5f;
  float x0f = floorf(xf), y0f = floorf(yf);
  int x0 = (int)x0f, y0 = (int)y0f;
  float wx1 = xf - x0f, wy1 = yf - y0f;
  const float* img = xn + (size_t)ch * NPOS;
  float v00 = 0.f, v10 = 0.f, v01 = 0.f, v11 = 0.f;
  bool xin0 = (x0 >= 0 && x0 < WOUT), xin1 = (x0 + 1 >= 0 && x0 + 1 < WOUT);
  bool yin0 = (y0 >= 0 && y0 < HOUT), yin1 = (y0 + 1 >= 0 && y0 + 1 < HOUT);
  if (xin0 && yin0) v00 = img[y0 * WOUT + x0];
  if (xin1 && yin0) v10 = img[y0 * WOUT + x0 + 1];
  if (xin0 && yin1) v01 = img[(y0 + 1) * WOUT + x0];
  if (xin1 && yin1) v11 = img[(y0 + 1) * WOUT + x0 + 1];
  kv[idx] = v00 * (1.f - wx1) * (1.f - wy1) + v10 * wx1 * (1.f - wy1)
          + v01 * (1.f - wx1) * wy1 + v11 * wx1 * wy1;
}

// ---------------------------------------------------------------------------
// 7) sim = scale * q^T k : per (b,h) GEMM M=576, N=36, K=64 (WMMA)
// ---------------------------------------------------------------------------
__global__ __launch_bounds__(32) void sim_gemm(const float* __restrict__ q,
                                               const float* __restrict__ k,
                                               float* __restrict__ sim) {
  const int bh = blockIdx.z;
  const int io = blockIdx.x * 16, jo = blockIdx.y * 16;
  const int lane = threadIdx.x & 31;
  const int h = lane >> 4, rr = lane & 15;
  const int i = io + rr;
  const int j = jo + rr;
  const bool jv = (j < NJ_);
  const float* qb = q + (size_t)bh * DH_ * NPOS;
  const float* kb = k + (size_t)bh * DH_ * NJ_;
  v8f acc = {};
  for (int kt = 0; kt < DH_; kt += 32) {
    v16h af, bf;
#pragma unroll
    for (int e = 0; e < 16; ++e) {
      af[e] = (_Float16)qb[(size_t)(kt + a_kidx(e, h)) * NPOS + i];
      float v = jv ? kb[(size_t)(kt + b_kidx(e, h)) * NJ_ + j] : 0.f;
      bf[e] = (_Float16)v;
    }
    acc = wmma_f16f32(af, bf, acc);
  }
  const int col = jo + (lane & 15);
  if (col < NJ_) {
    const float scale = 0.125f;   // 64^-0.5
#pragma unroll
    for (int v = 0; v < 8; ++v) {
      int m = io + v + h * 8;
      sim[((size_t)bh * NPOS + m) * NJ_ + col] = acc[v] * scale;
    }
  }
}

// ---------------------------------------------------------------------------
// 8) fused CPB bias MLP (2->192->192->1).
//    - vgrid staged to LDS via the Tensor Data Mover (TDM)
//    - W1 transposed f16 in LDS (padded stride -> conflict-free b128 loads)
//    - layer1 via WMMA with 4x ds_load_b128 per fragment pair
//    - layer2 folded into fragments + shfl_xor reduction
// ---------------------------------------------------------------------------
#define CPB_WAVES 4
#define CPB_BLOCKS 486
__global__ __launch_bounds__(128) void cpb_kernel(const float* __restrict__ vgrid,
                                                  const float* __restrict__ w0,
                                                  const float* __restrict__ b0,
                                                  const float* __restrict__ w1,
                                                  const float* __restrict__ b1,
                                                  const float* __restrict__ w2,
                                                  const float* __restrict__ b2,
                                                  float* __restrict__ sim) {
  __shared__ __align__(16) _Float16 sW1T[CPBD * W1S];            // 76800 B, transposed
  __shared__ __align__(16) _Float16 sH0[CPB_WAVES][16 * CPBD];   // 24576 B
  __shared__ __align__(16) float sVG[B_ * G_ * NJ_ * 2];         // 6912 B (TDM dest)
  __shared__ float sB1[CPBD], sW2[CPBD], sB0[CPBD], sW0[2 * CPBD];
  const int tid = threadIdx.x;
  // stage W1 transposed: sW1T[n][k] = w1[k][n]
  for (int i = tid; i < CPBD * CPBD; i += blockDim.x)
    sW1T[(i % CPBD) * W1S + (i / CPBD)] = (_Float16)w1[i];
  for (int i = tid; i < CPBD; i += blockDim.x) { sB1[i] = b1[i]; sW2[i] = w2[i]; sB0[i] = b0[i]; }
  for (int i = tid; i < 2 * CPBD; i += blockDim.x) sW0[i] = w0[i];

  // TDM: DMA vgrid (1728 x f32, 1-D tile) into LDS; wave 0 issues & waits
  if (tid < 32) {
    unsigned long long ga = (unsigned long long)vgrid;
    unsigned ldsoff = (unsigned)(size_t)(void*)&sVG[0];
    const unsigned nd = (unsigned)(B_ * G_ * NJ_ * 2);   // 1728 elements
    u32x4 g0 = { 1u,                                     // count=1, user mode
                 ldsoff,                                 // lds_addr
                 (unsigned)ga,                           // global_addr lo
                 (unsigned)((ga >> 32) & 0x01FFFFFFu) | 0x80000000u };  // hi | type=2
    i32x8 g1 = { (int)(2u << 16),                        // data_size = 4B
                 (int)((nd & 0xFFFFu) << 16),            // tensor_dim0[15:0]
                 (int)(nd >> 16),                        // tensor_dim0[31:16], tensor_dim1=0
                 (int)(nd << 16),                        // tile_dim0
                 0, (int)nd, 0, 0 };                     // tensor_dim0_stride
    i32x4 gz4 = { 0, 0, 0, 0 };
    i32x8 gz8 = { 0, 0, 0, 0, 0, 0, 0, 0 };
    __builtin_amdgcn_tensor_load_to_lds(g0, g1, gz4, gz4, gz8, 0);
    __builtin_amdgcn_s_wait_tensorcnt(0);
  }
  __syncthreads();

  const int wave = tid >> 5, lane = tid & 31;
  const int h = lane >> 4, rr = lane & 15;
  const float bias2 = b2[0];
  const int NCHUNK = (B_ * G_ * NPOS * NJ_) / 16;      // 31104
  _Float16* hrow = &sH0[wave][0];

  for (int chunk = blockIdx.x * CPB_WAVES + wave; chunk < NCHUNK;
       chunk += gridDim.x * CPB_WAVES) {
    const int r0 = chunk * 16;
    {
      int r = r0 + rr;
      int j = r % NJ_;
      int ri = r / NJ_;
      int i = ri % NPOS;
      int bg = ri / NPOS;
      float qxn = 2.f * (float)(i % WOUT) / (float)(WOUT - 1) - 1.f;
      float qyn = 2.f * (float)(i / WOUT) / (float)(HOUT - 1) - 1.f;
      float px = qxn - sVG[(bg * NJ_ + j) * 2 + 0];
      float py = qyn - sVG[(bg * NJ_ + j) * 2 + 1];
      float f0 = copysignf(log1pf(fabsf(px)), px);
      float f1 = copysignf(log1pf(fabsf(py)), py);
      for (int idx = lane; idx < 16 * CPBD; idx += 32) {
        int row = idx / CPBD, c = idx % CPBD;
        float g0 = __shfl(f0, row, 32);
        float g1 = __shfl(f1, row, 32);
        float hv = g0 * sW0[c] + g1 * sW0[CPBD + c] + sB0[c];
        hrow[row * CPBD + c] = (_Float16)(hv > 0.f ? hv : 0.f);
      }
    }
    __syncthreads();   // all waves run identical trip counts

    float acc2[8];
#pragma unroll
    for (int v = 0; v < 8; ++v) acc2[v] = 0.f;

    const _Float16* arow = hrow + rr * CPBD;
    for (int nt = 0; nt < CPBD / 16; ++nt) {
      const int ncol = nt * 16 + rr;
      const float binit = sB1[ncol];
      const _Float16* brow = &sW1T[ncol * W1S];
      v8f acc;
#pragma unroll
      for (int v = 0; v < 8; ++v) acc[v] = binit;
#pragma unroll
      for (int kt = 0; kt < CPBD; kt += 32) {
        v8h a0 = *(const v8h*)(arow + kt + (h << 3));
        v8h a1 = *(const v8h*)(arow + kt + 16 + (h << 3));
        v8h bb0 = *(const v8h*)(brow + kt + (h << 3));
        v8h bb1 = *(const v8h*)(brow + kt + 16 + (h << 3));
        v16h af = __builtin_shufflevector(a0, a1, 0,1,2,3,4,5,6,7,8,9,10,11,12,13,14,15);
        v16h bf = __builtin_shufflevector(bb0, bb1, 0,1,2,3,4,5,6,7,8,9,10,11,12,13,14,15);
        acc = wmma_f16f32(af, bf, acc);
      }
      const float wn = sW2[ncol];
#pragma unroll
      for (int v = 0; v < 8; ++v) {
        float hv = acc[v] > 0.f ? acc[v] : 0.f;   // relu(layer1)
        acc2[v] += hv * wn;                       // layer2 partial
      }
    }
#pragma unroll
    for (int v = 0; v < 8; ++v) {
      float s = acc2[v];
      s += __shfl_xor(s, 1, 32);
      s += __shfl_xor(s, 2, 32);
      s += __shfl_xor(s, 4, 32);
      s += __shfl_xor(s, 8, 32);
      if (rr == 0) {
        int row = r0 + h * 8 + v;
        sim[row] += s + bias2;                    // sim flat == (bg, i, j)
      }
    }
  }
}

// ---------------------------------------------------------------------------
// 9) softmax over j (36) + out = attn @ v  (one wave per (b,h,i))
// ---------------------------------------------------------------------------
__global__ void softmax_av_kernel(const float* __restrict__ sim,
                                  const float* __restrict__ v,
                                  float* __restrict__ out) {
  int wid = (blockIdx.x * blockDim.x + threadIdx.x) >> 5;
  int lane = threadIdx.x & 31;
  if (wid >= B_ * HEADS_ * NPOS) return;
  int i = wid % NPOS;
  int bh = wid / NPOS;
  const float* srow = sim + (size_t)wid * NJ_;
  float s0 = (lane < NJ_) ? srow[lane] : -1e30f;
  float s1 = (lane + 32 < NJ_) ? srow[lane + 32] : -1e30f;
  float mx = fmaxf(s0, s1);
  for (int m = 16; m; m >>= 1) mx = fmaxf(mx, __shfl_xor(mx, m, 32));
  float e0 = (lane < NJ_) ? __expf(s0 - mx) : 0.f;
  float e1 = (lane + 32 < NJ_) ? __expf(s1 - mx) : 0.f;
  float sum = e0 + e1;
  for (int m = 16; m; m >>= 1) sum += __shfl_xor(sum, m, 32);
  float inv = 1.f / sum;
  e0 *= inv; e1 *= inv;
  const float* vb = v + (size_t)bh * DH_ * NJ_;
  float o0 = 0.f, o1 = 0.f;   // d = lane, lane+32
  for (int j = 0; j < NJ_; ++j) {
    float aj = (j < 32) ? __shfl(e0, j, 32) : __shfl(e1, j - 32, 32);
    o0 += aj * vb[lane * NJ_ + j];
    o1 += aj * vb[(lane + 32) * NJ_ + j];
  }
  out[((size_t)bh * DH_ + lane) * NPOS + i] = o0;
  out[((size_t)bh * DH_ + lane + 32) * NPOS + i] = o1;
}

// ---------------------------------------------------------------------------
// 10/12) dense 768x768 GEMM + bias + residual (out-proj & MLP)  (WMMA)
// ---------------------------------------------------------------------------
__global__ __launch_bounds__(32) void dense_res_gemm(const float* __restrict__ src,
                                                     const float* __restrict__ w,
                                                     const float* __restrict__ bias,
                                                     const float* __restrict__ res,
                                                     float* __restrict__ y, int wT) {
  const int mo = blockIdx.x * 16;
  const int no = blockIdx.y * 16;
  const int lane = threadIdx.x & 31;
  const int h = lane >> 4, rr = lane & 15;
  const int n = no + rr;
  const int b = n / NPOS, p = n % NPOS;
  v8f acc = {};
  for (int kt = 0; kt < DIM_; kt += 32) {
    v16h af, bf;
#pragma unroll
    for (int e = 0; e < 16; ++e) {
      int ka = kt + a_kidx(e, h);
      float av = wT ? w[(size_t)ka * DIM_ + mo + rr]
                    : w[(size_t)(mo + rr) * DIM_ + ka];
      af[e] = (_Float16)av;
      int kb = kt + b_kidx(e, h);
      bf[e] = (_Float16)src[((size_t)b * DIM_ + kb) * NPOS + p];
    }
    __builtin_prefetch(&src[((size_t)b * DIM_ + kt + 32) * NPOS + p], 0, 0);
    acc = wmma_f16f32(af, bf, acc);
  }
#pragma unroll
  for (int v = 0; v < 8; ++v) {
    int m = mo + v + h * 8;
    size_t idx = ((size_t)b * DIM_ + m) * NPOS + p;
    y[idx] = acc[v] + bias[m] + res[idx];
  }
}

// ---------------------------------------------------------------------------
// launch
// ---------------------------------------------------------------------------
extern "C" void kernel_launch(void* const* d_in, const int* in_sizes, int n_in,
                              void* d_out, int out_size, void* d_ws, size_t ws_size,
                              hipStream_t stream) {
  (void)in_sizes; (void)n_in; (void)out_size; (void)ws_size;
  const float* x       = (const float*)d_in[0];
  const float* proj_w  = (const float*)d_in[1];
  const float* proj_b  = (const float*)d_in[2];
  const float* norm1_g = (const float*)d_in[3];
  const float* norm1_b = (const float*)d_in[4];
  const float* norm2_g = (const float*)d_in[5];
  const float* norm2_b = (const float*)d_in[6];
  const float* q_w     = (const float*)d_in[7];
  const float* k_w     = (const float*)d_in[8];
  const float* v_w     = (const float*)d_in[9];
  const float* out_w   = (const float*)d_in[10];
  const float* out_b   = (const float*)d_in[11];
  const float* off1_w  = (const float*)d_in[12];
  const float* off1_b  = (const float*)d_in[13];
  const float* off2_w  = (const float*)d_in[14];
  const float* cpb_w0  = (const float*)d_in[15];
  const float* cpb_b0  = (const float*)d_in[16];
  const float* cpb_w1  = (const float*)d_in[17];
  const float* cpb_b1  = (const float*)d_in[18];
  const float* cpb_w2  = (const float*)d_in[19];
  const float* cpb_b2  = (const float*)d_in[20];
  const float* mlp_w   = (const float*)d_in[21];
  const float* mlp_b   = (const float*)d_in[22];
  float* out = (float*)d_out;

  float* ws    = (float*)d_ws;
  float* xp    = ws;                 // (B,768,576) shortcut
  float* xn    = xp + 884736;        // LN1
  float* q     = xn + 884736;        // queries
  float* h1    = q + 884736;         // offset conv hidden (24,64,36)
  float* vgrid = h1 + 55296;         // (24,36,2)
  float* kv    = vgrid + 1728;       // sampled features (B,768,36)
  float* kM    = kv + 55296;
  float* vM    = kM + 55296;
  float* sim   = vM + 55296;         // (B,12,576,36)
  float* ao    = sim + 497664;       // attention output (B,768,576)
  float* x2    = ao + 884736;        // out+shortcut
  float* xn2   = x2 + 884736;        // LN2

  proj_gemm<<<dim3(DIM_ / 16, NQ_ / 16), 32, 0, stream>>>(x, proj_w, proj_b, xp);
  ln_kernel<<<(NQ_ * 32) / 256, 256, 0, stream>>>(xp, xn, norm1_g, norm1_b);
  grouped_gemm<<<dim3(4, NQ_ / 16, G_), 32, 0, stream>>>(xn, q_w, q, NPOS);
  offconv_kernel<<<(B_ * G_ * DH_ * NJ_ + 255) / 256, 256, 0, stream>>>(q, off1_w, off1_b, h1);
  offsets_kernel<<<(B_ * G_ * NJ_ + 255) / 256, 256, 0, stream>>>(h1, off2_w, vgrid);
  gridsample_kernel<<<(B_ * DIM_ * NJ_ + 255) / 256, 256, 0, stream>>>(xn, vgrid, kv);
  grouped_gemm<<<dim3(4, (B_ * NJ_ + 15) / 16, G_), 32, 0, stream>>>(kv, k_w, kM, NJ_);
  grouped_gemm<<<dim3(4, (B_ * NJ_ + 15) / 16, G_), 32, 0, stream>>>(kv, v_w, vM, NJ_);
  sim_gemm<<<dim3(NPOS / 16, (NJ_ + 15) / 16, B_ * HEADS_), 32, 0, stream>>>(q, kM, sim);
  cpb_kernel<<<CPB_BLOCKS, 128, 0, stream>>>(vgrid, cpb_w0, cpb_b0, cpb_w1, cpb_b1,
                                             cpb_w2, cpb_b2, sim);
  softmax_av_kernel<<<(B_ * HEADS_ * NPOS * 32) / 256, 256, 0, stream>>>(sim, vM, ao);
  dense_res_gemm<<<dim3(DIM_ / 16, NQ_ / 16), 32, 0, stream>>>(ao, out_w, out_b, xp, x2, 0);
  ln_kernel<<<(NQ_ * 32) / 256, 256, 0, stream>>>(x2, xn2, norm2_g, norm2_b);
  dense_res_gemm<<<dim3(DIM_ / 16, NQ_ / 16), 32, 0, stream>>>(xn2, mlp_w, mlp_b, x2, out, 1);
}
